// UniqueEq3Net_20349555049142
// MI455X (gfx1250) — matrix operand order repaired and checked by reading, not compiled, not run on hardware
//
#include <hip/hip_runtime.h>
#include <hip/hip_bf16.h>

// MI455X (gfx1250) implementation.
// Heavy math: per-(b,j,k) GEMMs E(40x64)*B(64x64) and y(40x64)*V0(64x32)
// via v_wmma_f32_16x16x32_bf16 (wave32, 16x16 tiles, f32 accumulate).
// All WMMA B-operands are staged COLUMN-major in LDS so a lane's fragment is
// one contiguous 32-byte region -> 2x ds_load_b128 instead of 16x ds_load_u16.
// Workspace requirement: ~45 MB (see layout below).

typedef __attribute__((ext_vector_type(16))) __bf16 v16bf;
typedef __attribute__((ext_vector_type(8)))  __bf16 v8bf;
typedef __attribute__((ext_vector_type(2)))  __bf16 v2bf;
typedef __attribute__((ext_vector_type(8)))  float  v8f;

#define NB   16
#define NK   40
#define ND   64
#define NO2  32

// ---------------- workspace layout (float offsets) ----------------
#define OFF_E      0u          // 16*40*64        = 40960
#define OFF_S      40960u      // 16*64           = 1024
// zeroed-every-call span starts here (atomic targets / accumulators)
#define OFF_YJ     41984u      // 16*1600*64      = 1638400  (atomic)
#define OFF_YIJ    1680384u    // 16*40*64        = 40960    (atomic)
#define OFF_YJK    1721344u    // 16*40*64        = 40960    (atomic)
#define OFF_YTOT   1762304u    // 16*64           = 1024     (atomic)
#define OFF_OUTACC 1763328u    // 16*32           = 512      (atomic)
#define ZERO_BASE  OFF_YJ
#define ZERO_COUNT 1721856u
// direct-write buffers
#define OFF_YI     1763840u    // 16*1600*64
#define OFF_YK     3402240u    // 16*1600*64
#define OFF_YIK    5040640u    // 16*40*64
#define OFF_G1P    5081600u    // 16*1600*32
#define OFF_G2P    5900800u
#define OFF_G3P    6720000u
#define OFF_R4     7539200u    // 16*40*32
#define OFF_R5     7559680u
#define OFF_R6     7580160u
#define OFF_C2     7600640u    // 16*32
#define OFF_C1BUF  7601152u    // 640*32*12*12 = 2949120
#define OFF_C2BUF  10550272u   // 640*1024     = 655360
// total = 11205632 floats = 44.8 MB

// ---------------- utility ----------------
__global__ void k_zero(float* p, unsigned n) {
    unsigned g = blockIdx.x * blockDim.x + threadIdx.x;
    if (g < n) p[g] = 0.0f;
}

// ---------------- CNN front end ----------------
__global__ void k_conv1(const float* __restrict__ x, const float* __restrict__ w1,
                        const float* __restrict__ b1, float* __restrict__ c1buf) {
    unsigned g = blockIdx.x * blockDim.x + threadIdx.x;
    if (g >= 640u * 32u * 144u) return;
    int px = g % 12, py = (g / 12) % 12, oc = (g / 144) % 32, img = g / (144 * 32);
    const float* xi = x + (size_t)img * 784;
    const float* wf = w1 + oc * 25;
    float best = -1e30f;
    for (int dy = 0; dy < 2; ++dy)
      for (int dx = 0; dx < 2; ++dx) {
        int r0 = 2 * py + dy, c0 = 2 * px + dx;
        float acc = b1[oc];
#pragma unroll
        for (int u = 0; u < 5; ++u)
#pragma unroll
          for (int v = 0; v < 5; ++v)
            acc += xi[(r0 + u) * 28 + (c0 + v)] * wf[u * 5 + v];
        best = fmaxf(best, acc);
      }
    c1buf[g] = fmaxf(best, 0.0f);
}

__global__ __launch_bounds__(128) void k_conv2(const float* __restrict__ c1buf,
                                               const float* __restrict__ w2,
                                               const float* __restrict__ b2,
                                               float* __restrict__ c2buf) {
    __shared__ float in_s[32 * 144];
    int img = blockIdx.x, t = threadIdx.x;
    for (int idx = t; idx < 4608; idx += 128) in_s[idx] = c1buf[(size_t)img * 4608 + idx];
    __syncthreads();
    for (int o = t; o < 1024; o += 128) {
        int oc = o >> 4, py = (o >> 2) & 3, px = o & 3;
        float best = -1e30f;
        for (int dy = 0; dy < 2; ++dy)
          for (int dx = 0; dx < 2; ++dx) {
            int r0 = 2 * py + dy, c0 = 2 * px + dx;
            float acc = b2[oc];
            for (int ic = 0; ic < 32; ++ic) {
                const float* ip = &in_s[ic * 144 + r0 * 12 + c0];
                const float* wp = &w2[(size_t)oc * 800 + ic * 25];
#pragma unroll
                for (int u = 0; u < 5; ++u)
#pragma unroll
                  for (int v = 0; v < 5; ++v)
                    acc += ip[u * 12 + v] * wp[u * 5 + v];
            }
            best = fmaxf(best, acc);
          }
        c2buf[(size_t)img * 1024 + o] = fmaxf(best, 0.0f);
    }
}

__global__ void k_fc(const float* __restrict__ c2buf, const float* __restrict__ fcw,
                     const float* __restrict__ fcb, float* __restrict__ e_ws) {
    unsigned g = blockIdx.x * blockDim.x + threadIdx.x;
    if (g >= 640u * 64u) return;
    int h = g & 63, img = g >> 6;
    float acc = fcb[h];
    const float* zp = c2buf + (size_t)img * 1024;
#pragma unroll 8
    for (int f = 0; f < 1024; ++f) acc += zp[f] * fcw[(size_t)f * 64 + h];
    e_ws[(size_t)img * 64 + h] = fmaxf(acc, 0.0f);
}

__global__ void k_colsum(const float* __restrict__ e_ws, float* __restrict__ S_ws) {
    unsigned g = blockIdx.x * blockDim.x + threadIdx.x;
    if (g >= NB * ND) return;
    int b = g >> 6, d = g & 63;
    float s = 0.0f;
    for (int i = 0; i < NK; ++i) s += e_ws[((size_t)b * NK + i) * ND + d];
    S_ws[g] = s;
}

// ---------------- eq3to3 pass B ----------------
__global__ __launch_bounds__(128) void k_passB(
    const float* __restrict__ e_ws, const float* __restrict__ S_ws,
    const float* __restrict__ W, const float* __restrict__ b1,
    float* __restrict__ Yi, float* __restrict__ Yj, float* __restrict__ Yk,
    float* __restrict__ Yij, float* __restrict__ Yik, float* __restrict__ Yjk,
    float* __restrict__ Ytot) {
    __shared__ float e_s[NK * ND], S_s[ND];
    __shared__ float DmT[ND * 65], CmT[ND * 65];   // [o][d], padded stride 65
    __shared__ float Fm[ND * ND], g_s[ND], r_s[ND];
    __shared__ __bf16 Bcol[ND * ND];               // COLUMN major: [n*64 + K]
    __shared__ float yst[NK * ND], accK[NK * ND];

    int wg = blockIdx.x, b = wg / NK, j = wg % NK, t = threadIdx.x;
    for (int idx = t; idx < NK * ND; idx += 128) {
        e_s[idx]  = e_ws[((size_t)b * NK) * ND + idx];
        accK[idx] = 0.0f;
    }
    if (t < ND) S_s[t] = S_ws[b * ND + t];
    __syncthreads();

    // per-WG operand matrices (coalesced W reads; transposed+padded LDS writes)
    for (int idx = t; idx < ND * ND; idx += 128) {
        int d = idx >> 6, o = idx & 63;
        float S1 = S_s[d], S2 = S1 * S1, ej = e_s[j * ND + d];
        float w0 = W[idx], w1 = W[4096 + idx], w2 = W[2 * 4096 + idx];
        float w3 = W[3 * 4096 + idx], w4 = W[4 * 4096 + idx], w6 = W[6 * 4096 + idx];
        DmT[o * 65 + d] = ej * w0 + S1 * w2;
        CmT[o * 65 + d] = ej * S1 * w3 + S2 * w6;
        Fm[idx]         = ej * S1 * w1 + S2 * w4;
    }
    if (t < ND) {
        float acc = b1[t];
        for (int d = 0; d < ND; ++d) {
            float S1 = S_s[d], S2 = S1 * S1;
            acc += e_s[j * ND + d] * S2 * W[5 * 4096 + d * 64 + t]
                 + S2 * S1 * W[7 * 4096 + d * 64 + t];
        }
        g_s[t] = acc;
    }
    __syncthreads();

    int lane = t & 31, w = t >> 5;
    int m = lane & 15, kh = lane >> 4;
    // A frags from E (constant over whole k loop) — contiguous k order (k = 16*kh + elem), matching Bcol B frags
    v16bf afrag[3][2];
#pragma unroll
    for (int mt = 0; mt < 3; ++mt) {
        int row = mt * 16 + m;
#pragma unroll
        for (int s = 0; s < 2; ++s) {
            v16bf a;
#pragma unroll
            for (int v = 0; v < 8; ++v) {
                int kk = s * 32 + kh * 16 + 2 * v;
                float x0 = (row < NK) ? e_s[row * ND + kk] : 0.0f;
                float x1 = (row < NK) ? e_s[row * ND + kk + 1] : 0.0f;
                a[2 * v] = (__bf16)x0; a[2 * v + 1] = (__bf16)x1;
            }
            afrag[mt][s] = a;
        }
    }

    float accYik = 0.0f;
    int nt = w;
    int nlane = nt * 16 + (lane & 15);
    int kb = (lane >= 16) ? 16 : 0;
    for (int k = 0; k < NK; ++k) {
        // Bcol[n][d] = e_k[d]*D[d][n] + C[d][n], packed pair stores
        for (int idx2 = t; idx2 < 2048; idx2 += 128) {
            int n = idx2 >> 5, d0 = (idx2 & 31) * 2;
            float v0 = e_s[k * ND + d0]     * DmT[n * 65 + d0]     + CmT[n * 65 + d0];
            float v1 = e_s[k * ND + d0 + 1] * DmT[n * 65 + d0 + 1] + CmT[n * 65 + d0 + 1];
            v2bf pr; pr[0] = (__bf16)v0; pr[1] = (__bf16)v1;
            *(v2bf*)&Bcol[n * 64 + d0] = pr;
        }
        if (t < ND) {
            float acc = g_s[t];
#pragma unroll 8
            for (int d = 0; d < ND; ++d) acc += e_s[k * ND + d] * Fm[d * 64 + t];
            r_s[t] = acc;
        }
        __syncthreads();

        // B frags: contiguous 32B in Bcol -> 2x ds_load_b128 each
        v16bf bfrag0 = *(const v16bf*)&Bcol[nlane * 64 + kb];
        v16bf bfrag1 = *(const v16bf*)&Bcol[nlane * 64 + 32 + kb];
#pragma unroll
        for (int mt = 0; mt < 3; ++mt) {
            v8f acc = {};
            acc = __builtin_amdgcn_wmma_f32_16x16x32_bf16(false, afrag[mt][0], false, bfrag0, (short)0, acc, false, false);
            acc = __builtin_amdgcn_wmma_f32_16x16x32_bf16(false, afrag[mt][1], false, bfrag1, (short)0, acc, false, false);
            int rbase = (lane >= 16) ? 8 : 0;
#pragma unroll
            for (int rr = 0; rr < 8; ++rr) {
                int row = mt * 16 + rbase + rr;
                if (row < NK) {
                    float vv = acc[rr] + r_s[nlane];
                    yst[row * ND + nlane] = vv > 0.0f ? vv : 0.0f;
                }
            }
        }
        __syncthreads();

        if (t < ND) {  // Yi (direct), Yij (atomic across j), Yik partial
            float s = 0.0f;
            for (int i = 0; i < NK; ++i) s += yst[i * ND + t];
            Yi[((size_t)b * 1600 + j * NK + k) * ND + t] = s;
            atomicAdd(&Yij[((size_t)b * NK + k) * ND + t], s);
            accYik += s;
        }
        for (int idx = t; idx < NK * ND; idx += 128) {  // Yj (atomic), Yk partial
            float vv = yst[idx];
            accK[idx] += vv;
            int i = idx >> 6, o = idx & 63;
            atomicAdd(&Yj[((size_t)b * 1600 + i * NK + k) * ND + o], vv);
        }
        __syncthreads();
    }
    for (int idx = t; idx < NK * ND; idx += 128) {
        int i = idx >> 6, o = idx & 63;
        Yk[((size_t)b * 1600 + i * NK + j) * ND + o] = accK[idx];
        atomicAdd(&Yjk[((size_t)b * NK + i) * ND + o], accK[idx]);
    }
    if (t < ND) {
        Yik[((size_t)b * NK + j) * ND + t] = accYik;
        atomicAdd(&Ytot[b * ND + t], accYik);
    }
}

// ---------------- pass B2: broadcast-term precompute ----------------
__global__ void k_passB2(const float* __restrict__ Yi, const float* __restrict__ Yj,
                         const float* __restrict__ Yk, const float* __restrict__ Yij,
                         const float* __restrict__ Yik, const float* __restrict__ Yjk,
                         const float* __restrict__ Ytot, const float* __restrict__ V,
                         const float* __restrict__ b2, float* __restrict__ G1p,
                         float* __restrict__ G2p, float* __restrict__ G3p,
                         float* __restrict__ R4, float* __restrict__ R5,
                         float* __restrict__ R6, float* __restrict__ c2) {
    const unsigned PER_B = 3u * 51200u + 3u * 1280u + 32u;  // 157472
    unsigned g = blockIdx.x * blockDim.x + threadIdx.x;
    if (g >= NB * PER_B) return;
    int b = g / PER_B; unsigned r = g % PER_B;
    const float* src; const float* Vop; float* dst;
    if (r < 51200u)        { src = Yi + (size_t)b * 102400; Vop = V + 2048;     dst = G1p + (size_t)b * 51200; }
    else if (r < 102400u)  { r -= 51200u;  src = Yj + (size_t)b * 102400; Vop = V + 2 * 2048; dst = G2p + (size_t)b * 51200; }
    else if (r < 153600u)  { r -= 102400u; src = Yk + (size_t)b * 102400; Vop = V + 3 * 2048; dst = G3p + (size_t)b * 51200; }
    else if (r < 154880u)  { r -= 153600u; src = Yij + (size_t)b * 2560; Vop = V + 4 * 2048; dst = R4 + (size_t)b * 1280; }
    else if (r < 156160u)  { r -= 154880u; src = Yik + (size_t)b * 2560; Vop = V + 5 * 2048; dst = R5 + (size_t)b * 1280; }
    else if (r < 157440u)  { r -= 156160u; src = Yjk + (size_t)b * 2560; Vop = V + 6 * 2048; dst = R6 + (size_t)b * 1280; }
    else {
        r -= 157440u;
        float acc = b2[r];
        for (int d = 0; d < ND; ++d) acc += Ytot[b * ND + d] * V[7 * 2048 + d * 32 + r];
        c2[b * 32 + r] = acc;
        return;
    }
    int row = r >> 5, o = r & 31;
    float acc = 0.0f;
#pragma unroll 8
    for (int d = 0; d < ND; ++d) acc += src[(size_t)row * ND + d] * Vop[d * 32 + o];
    dst[(size_t)row * 32 + o] = acc;
}

// ---------------- eq3to3 pass C ----------------
__global__ __launch_bounds__(128) void k_passC(
    const float* __restrict__ e_ws, const float* __restrict__ S_ws,
    const float* __restrict__ W, const float* __restrict__ b1,
    const float* __restrict__ V, const float* __restrict__ G1p,
    const float* __restrict__ G2p, const float* __restrict__ G3p,
    const float* __restrict__ R4, const float* __restrict__ R5,
    const float* __restrict__ R6, const float* __restrict__ c2,
    float* __restrict__ outacc) {
    __shared__ float e_s[NK * ND], S_s[ND];
    __shared__ float DmT[ND * 65], CmT[ND * 65];
    __shared__ float Fm[ND * ND], g_s[ND], r_s[ND];
    __shared__ __bf16 Bcol[ND * ND];               // [n*64 + K]
    __shared__ __bf16 ystb[48 * ND];               // row major, rows 40..47 zero
    __shared__ __bf16 V0c[NO2 * ND];               // COLUMN major: [o2*64 + K]
    __shared__ float y2st[NK * NO2];
    __shared__ float part[128];

    int wg = blockIdx.x, b = wg / NK, j = wg % NK, t = threadIdx.x;
    for (int idx = t; idx < NK * ND; idx += 128) e_s[idx] = e_ws[((size_t)b * NK) * ND + idx];
    if (t < ND) S_s[t] = S_ws[b * ND + t];
    for (int idx = t; idx < NO2 * ND; idx += 128) {
        int o2 = idx >> 6, d = idx & 63;
        V0c[idx] = (__bf16)V[d * 32 + o2];
    }
    for (int idx = NK * ND + t; idx < 48 * ND; idx += 128) ystb[idx] = (__bf16)0.0f;
    __syncthreads();

    for (int idx = t; idx < ND * ND; idx += 128) {
        int d = idx >> 6, o = idx & 63;
        float S1 = S_s[d], S2 = S1 * S1, ej = e_s[j * ND + d];
        float w0 = W[idx], w1 = W[4096 + idx], w2 = W[2 * 4096 + idx];
        float w3 = W[3 * 4096 + idx], w4 = W[4 * 4096 + idx], w6 = W[6 * 4096 + idx];
        DmT[o * 65 + d] = ej * w0 + S1 * w2;
        CmT[o * 65 + d] = ej * S1 * w3 + S2 * w6;
        Fm[idx]         = ej * S1 * w1 + S2 * w4;
    }
    if (t < ND) {
        float acc = b1[t];
        for (int d = 0; d < ND; ++d) {
            float S1 = S_s[d], S2 = S1 * S1;
            acc += e_s[j * ND + d] * S2 * W[5 * 4096 + d * 64 + t]
                 + S2 * S1 * W[7 * 4096 + d * 64 + t];
        }
        g_s[t] = acc;
    }
    __syncthreads();

    int lane = t & 31, w = t >> 5;
    int m = lane & 15, kh = lane >> 4;
    v16bf afrag[3][2];
#pragma unroll
    for (int mt = 0; mt < 3; ++mt) {
        int row = mt * 16 + m;
#pragma unroll
        for (int s = 0; s < 2; ++s) {
            v16bf a;
#pragma unroll
            for (int v = 0; v < 8; ++v) {
                int kk = s * 32 + kh * 16 + 2 * v;
                float x0 = (row < NK) ? e_s[row * ND + kk] : 0.0f;
                float x1 = (row < NK) ? e_s[row * ND + kk + 1] : 0.0f;
                a[2 * v] = (__bf16)x0; a[2 * v + 1] = (__bf16)x1;
            }
            afrag[mt][s] = a;
        }
    }
    int kb = (lane >= 16) ? 16 : 0;
    // GEMM2 B frags (constant V0): waves 0,1 own the two 16-wide N tiles
    v16bf b2frag0, b2frag1;
    int n2 = (w & 1) * 16 + (lane & 15);
    if (w < 2) {
        b2frag0 = *(const v16bf*)&V0c[n2 * 64 + kb];
        b2frag1 = *(const v16bf*)&V0c[n2 * 64 + 32 + kb];
    }

    float partial = 0.0f;
    int nt = w;
    int nlane = nt * 16 + (lane & 15);
    for (int k = 0; k < NK; ++k) {
        for (int idx2 = t; idx2 < 2048; idx2 += 128) {
            int n = idx2 >> 5, d0 = (idx2 & 31) * 2;
            float v0 = e_s[k * ND + d0]     * DmT[n * 65 + d0]     + CmT[n * 65 + d0];
            float v1 = e_s[k * ND + d0 + 1] * DmT[n * 65 + d0 + 1] + CmT[n * 65 + d0 + 1];
            v2bf pr; pr[0] = (__bf16)v0; pr[1] = (__bf16)v1;
            *(v2bf*)&Bcol[n * 64 + d0] = pr;
        }
        if (t < ND) {
            float acc = g_s[t];
#pragma unroll 8
            for (int d = 0; d < ND; ++d) acc += e_s[k * ND + d] * Fm[d * 64 + t];
            r_s[t] = acc;
        }
        __syncthreads();

        v16bf bfrag0 = *(const v16bf*)&Bcol[nlane * 64 + kb];
        v16bf bfrag1 = *(const v16bf*)&Bcol[nlane * 64 + 32 + kb];
#pragma unroll
        for (int mt = 0; mt < 3; ++mt) {
            v8f acc = {};
            acc = __builtin_amdgcn_wmma_f32_16x16x32_bf16(false, afrag[mt][0], false, bfrag0, (short)0, acc, false, false);
            acc = __builtin_amdgcn_wmma_f32_16x16x32_bf16(false, afrag[mt][1], false, bfrag1, (short)0, acc, false, false);
            int rbase = (lane >= 16) ? 8 : 0;
#pragma unroll
            for (int rr = 0; rr < 8; ++rr) {
                int row = mt * 16 + rbase + rr;
                if (row < NK) {
                    float vv = acc[rr] + r_s[nlane];
                    ystb[row * ND + nlane] = (__bf16)(vv > 0.0f ? vv : 0.0f);
                }
            }
        }
        __syncthreads();

        if (w < 2) {  // GEMM2: y(48x64) * V0(64x32); A frags = 2x contiguous 16B halves
#pragma unroll
            for (int mt = 0; mt < 3; ++mt) {
                int row = mt * 16 + m;
                v8bf lo0 = *(const v8bf*)&ystb[row * ND + kh * 16];
                v8bf hi0 = *(const v8bf*)&ystb[row * ND + kh * 16 + 8];
                v8bf lo1 = *(const v8bf*)&ystb[row * ND + 32 + kh * 16];
                v8bf hi1 = *(const v8bf*)&ystb[row * ND + 32 + kh * 16 + 8];
                v16bf a0 = __builtin_shufflevector(lo0, hi0, 0,1,2,3,4,5,6,7,8,9,10,11,12,13,14,15);
                v16bf a1 = __builtin_shufflevector(lo1, hi1, 0,1,2,3,4,5,6,7,8,9,10,11,12,13,14,15);
                v8f acc2 = {};
                acc2 = __builtin_amdgcn_wmma_f32_16x16x32_bf16(false, a0, false, b2frag0, (short)0, acc2, false, false);
                acc2 = __builtin_amdgcn_wmma_f32_16x16x32_bf16(false, a1, false, b2frag1, (short)0, acc2, false, false);
                int rbase = (lane >= 16) ? 8 : 0;
#pragma unroll
                for (int rr = 0; rr < 8; ++rr) {
                    int row2 = mt * 16 + rbase + rr;
                    if (row2 < NK) y2st[row2 * NO2 + n2] = acc2[rr];
                }
            }
        }
        __syncthreads();

        for (int idx = t; idx < NK * NO2; idx += 128) {
            int i = idx >> 5, o = idx & 31;
            float vv = y2st[idx]
                     + G1p[((size_t)b * 1600 + j * NK + k) * 32 + o]
                     + G2p[((size_t)b * 1600 + i * NK + k) * 32 + o]
                     + G3p[((size_t)b * 1600 + i * NK + j) * 32 + o]
                     + R4[((size_t)b * NK + k) * 32 + o]
                     + R5[((size_t)b * NK + j) * 32 + o]
                     + R6[((size_t)b * NK + i) * 32 + o]
                     + c2[b * 32 + o];
            partial += vv > 0.0f ? vv : 0.0f;
        }
        __syncthreads();
    }
    part[t] = partial;
    __syncthreads();
    if (t < 32) atomicAdd(&outacc[b * 32 + t], part[t] + part[t + 32] + part[t + 64] + part[t + 96]);
}

// ---------------- final head ----------------
__global__ void k_out(const float* __restrict__ outacc, const float* __restrict__ outw,
                      const float* __restrict__ outb, float* __restrict__ out) {
    unsigned g = blockIdx.x * blockDim.x + threadIdx.x;
    if (g >= NB) return;
    float acc = outb[0];
    for (int o = 0; o < NO2; ++o) acc += fmaxf(outacc[g * NO2 + o], 0.0f) * outw[o];
    out[g] = acc;
}

extern "C" void kernel_launch(void* const* d_in, const int* in_sizes, int n_in,
                              void* d_out, int out_size, void* d_ws, size_t ws_size,
                              hipStream_t stream) {
    const float* x    = (const float*)d_in[0];
    const float* w1   = (const float*)d_in[1];
    const float* b1c  = (const float*)d_in[2];
    const float* w2   = (const float*)d_in[3];
    const float* b2c  = (const float*)d_in[4];
    const float* fcw  = (const float*)d_in[5];
    const float* fcb  = (const float*)d_in[6];
    const float* eq1w = (const float*)d_in[7];
    const float* eq1b = (const float*)d_in[8];
    const float* eq2w = (const float*)d_in[9];
    const float* eq2b = (const float*)d_in[10];
    const float* outw = (const float*)d_in[11];
    const float* outb = (const float*)d_in[12];
    float* ws = (float*)d_ws;

    float* e_ws  = ws + OFF_E;   float* S_ws  = ws + OFF_S;
    float* Yi    = ws + OFF_YI;  float* Yj    = ws + OFF_YJ;   float* Yk  = ws + OFF_YK;
    float* Yij   = ws + OFF_YIJ; float* Yik   = ws + OFF_YIK;  float* Yjk = ws + OFF_YJK;
    float* Ytot  = ws + OFF_YTOT; float* oac  = ws + OFF_OUTACC;
    float* G1p   = ws + OFF_G1P; float* G2p   = ws + OFF_G2P;  float* G3p = ws + OFF_G3P;
    float* R4    = ws + OFF_R4;  float* R5    = ws + OFF_R5;   float* R6  = ws + OFF_R6;
    float* c2    = ws + OFF_C2;
    float* c1buf = ws + OFF_C1BUF; float* c2buf = ws + OFF_C2BUF;

    k_zero<<<(ZERO_COUNT + 255) / 256, 256, 0, stream>>>(ws + ZERO_BASE, ZERO_COUNT);
    k_conv1<<<(640u * 32u * 144u + 255) / 256, 256, 0, stream>>>(x, w1, b1c, c1buf);
    k_conv2<<<640, 128, 0, stream>>>(c1buf, w2, b2c, c2buf);
    k_fc<<<(640u * 64u + 255) / 256, 256, 0, stream>>>(c2buf, fcw, fcb, e_ws);
    k_colsum<<<(NB * ND + 255) / 256, 256, 0, stream>>>(e_ws, S_ws);
    k_passB<<<NB * NK, 128, 0, stream>>>(e_ws, S_ws, eq1w, eq1b, Yi, Yj, Yk, Yij, Yik, Yjk, Ytot);
    k_passB2<<<(NB * 157472u + 255) / 256, 256, 0, stream>>>(Yi, Yj, Yk, Yij, Yik, Yjk,
                                                            Ytot, eq2w, eq2b,
                                                            G1p, G2p, G3p, R4, R5, R6, c2);
    k_passC<<<NB * NK, 128, 0, stream>>>(e_ws, S_ws, eq1w, eq1b, eq2w,
                                         G1p, G2p, G3p, R4, R5, R6, c2, oac);
    k_out<<<1, 32, 0, stream>>>(oac, outw, outb, (float*)d_out);
}
